// ScaledDotProductAttention_36438502539806
// MI455X (gfx1250) — matrix operand order repaired and checked by reading, not compiled, not run on hardware
//
#include <hip/hip_runtime.h>

typedef __attribute__((ext_vector_type(16))) __bf16        v16bf;
typedef __attribute__((ext_vector_type(8)))  float         v8f;
typedef __attribute__((ext_vector_type(8)))  unsigned int  v8u;

#define WMMA_BF16(A, B, C)                                                     \
  __builtin_amdgcn_wmma_f32_16x16x32_bf16(                                     \
      false, __builtin_bit_cast(v16bf, (A)), false,                            \
      __builtin_bit_cast(v16bf, (B)), (short)0, (C), false, false)

static __device__ __forceinline__ unsigned short f2bf(float f) {
  unsigned int u = __builtin_bit_cast(unsigned int, f);
  u += 0x7FFFu + ((u >> 16) & 1u);  // RNE
  return (unsigned short)(u >> 16);
}
static __device__ __forceinline__ unsigned int pack2bf(float a, float b) {
  return (unsigned int)f2bf(a) | ((unsigned int)f2bf(b) << 16);
}

// Per-lane async copy of 16B global -> LDS (gfx1250 ASYNC path, ASYNCcnt).
static __device__ __forceinline__ void async_ld_b128(void* lds, const void* g) {
  unsigned lo = (unsigned)(size_t)lds;                 // low 32b of generic = LDS offset
  unsigned long long ga = (unsigned long long)(size_t)g;
  asm volatile("global_load_async_to_lds_b128 %0, %1, off" :: "v"(lo), "v"(ga) : "memory");
}

constexpr int kB = 2, kH = 16, kS = 2048, kD = 64;
constexpr int NW = 4;           // waves per block
constexpr int KC = 32;          // keys per chunk
constexpr float SCALE = 0.125f; // 1/sqrt(64)

// ---------------- main (ws) kernel: QT=32 rows/wave, bf16 K / V^T from ws ----
constexpr int QT2   = 32;
constexpr int KSTR2 = 72;  // K tile [key][d] stride  (144B rows: 16B-aligned segs, conflict-free)
constexpr int VSTR2 = 40;  // V^T tile [d][key] stride (80B rows)
constexpr int PSTR  = 34;  // P tile [q][key] stride
constexpr int KTILE = KC * KSTR2;   // shorts per K buffer
constexpr int VTILE = kD * VSTR2;   // shorts per V^T buffer

// Stage one 32-key chunk (K row-major + V^T) into an LDS buffer, async.
static __device__ __forceinline__ void stage_chunk(
    const unsigned short* Kp, const unsigned short* Vp,
    unsigned short* bK, unsigned short* bV, int kb, int tid) {
  const int krow = tid >> 2, kseg = (tid & 3) * 16;
  async_ld_b128(&bK[krow * KSTR2 + kseg],     Kp + (size_t)(kb + krow) * kD + kseg);
  async_ld_b128(&bK[krow * KSTR2 + kseg + 8], Kp + (size_t)(kb + krow) * kD + kseg + 8);
  const int d = tid >> 1, dh = (tid & 1) * 16;
  async_ld_b128(&bV[d * VSTR2 + dh],     Vp + (size_t)d * kS + kb + dh);
  async_ld_b128(&bV[d * VSTR2 + dh + 8], Vp + (size_t)d * kS + kb + dh + 8);
}

__global__ __launch_bounds__(NW * 32) void fa2_wmma_bf16_kernel(
    const float* __restrict__ Qg, const unsigned short* __restrict__ Kbf,
    const unsigned short* __restrict__ VTbf, float* __restrict__ Og) {
  __shared__ unsigned short sK[2 * KTILE];         // double-buffered K chunk [key][d]
  __shared__ unsigned short sVT[2 * VTILE];        // double-buffered V^T chunk [d][key]
  __shared__ unsigned short sP[NW * QT2 * PSTR];   // per-wave probs

  const int tid  = threadIdx.x;
  const int wave = tid >> 5;
  const int lane = tid & 31;
  const int ln   = lane & 15;
  const int hi   = lane >> 4;

  const int qblocks = kS / (QT2 * NW);  // 16
  const int bh = blockIdx.x / qblocks;
  const int qb = (blockIdx.x % qblocks) * (QT2 * NW);
  const int q0 = qb + wave * QT2;

  const float*          Qp = Qg   + (size_t)bh * kS * kD;
  const unsigned short* Kp = Kbf  + (size_t)bh * kS * kD;   // [s][d]
  const unsigned short* Vp = VTbf + (size_t)bh * kS * kD;   // [d][s]
  float*                Op = Og   + (size_t)bh * kS * kD;

  // Q -> A operands, pre-scaled: qa[q-subtile][d-half]
  v8u qa[2][2];
#pragma unroll
  for (int qs = 0; qs < 2; ++qs) {
    const float* qrow = Qp + (size_t)(q0 + qs * 16 + ln) * kD;
#pragma unroll
    for (int v = 0; v < 8; ++v) {
      const int k0 = 2 * v + (hi ? 16 : 0);
      float2 f0 = *(const float2*)(qrow + k0);
      float2 f1 = *(const float2*)(qrow + k0 + 32);
      qa[qs][0][v] = pack2bf(f0.x * SCALE, f0.y * SCALE);
      qa[qs][1][v] = pack2bf(f1.x * SCALE, f1.y * SCALE);
    }
  }

  v8f acc[2][4] = {};
  float m[2][8], l[2][8];
#pragma unroll
  for (int qs = 0; qs < 2; ++qs)
#pragma unroll
    for (int r = 0; r < 8; ++r) { m[qs][r] = -__builtin_inff(); l[qs][r] = 0.0f; }

  const int cmax = (qb + QT2 * NW - 1) / KC;

  // ---- software pipeline: prefetch chunk 0 ----
  stage_chunk(Kp, Vp, &sK[0], &sVT[0], 0, tid);

  for (int c = 0; c <= cmax; ++c) {
    const int kb = c * KC;
    unsigned short* cK = &sK[(c & 1) * KTILE];
    unsigned short* cV = &sVT[(c & 1) * VTILE];

    __syncthreads();  // A: all waves done reading the buffer we are about to overwrite
    if (c < cmax) {
      stage_chunk(Kp, Vp, &sK[((c + 1) & 1) * KTILE], &sVT[((c + 1) & 1) * VTILE],
                  kb + KC, tid);
      asm volatile("s_wait_asynccnt 4" ::: "memory");  // chunk c complete; c+1 in flight
    } else {
      asm volatile("s_wait_asynccnt 0" ::: "memory");
    }
    __syncthreads();  // B: every wave's part of chunk c landed in LDS

    if (kb <= q0 + QT2 - 1) {  // wave-uniform causal guard
      // ---- scores: 4 tiles (2 q-subtiles x 2 key-subtiles) ----
      v8f s[2][2];
#pragma unroll
      for (int sub = 0; sub < 2; ++sub) {
        v8u b0, b1;
        const int krow = sub * 16 + ln;
#pragma unroll
        for (int v = 0; v < 8; ++v) {
          const int d0 = 2 * v + (hi ? 16 : 0);
          b0[v] = *(const unsigned int*)&cK[krow * KSTR2 + d0];
          b1[v] = *(const unsigned int*)&cK[krow * KSTR2 + d0 + 32];
        }
#pragma unroll
        for (int qs = 0; qs < 2; ++qs) {
          v8f z = {};
          z = WMMA_BF16(qa[qs][0], b0, z);
          z = WMMA_BF16(qa[qs][1], b1, z);
          s[qs][sub] = z;
        }
      }

      // ---- causal mask (diagonal chunks) ----
      if (kb + KC > q0) {
#pragma unroll
        for (int qs = 0; qs < 2; ++qs)
#pragma unroll
          for (int sub = 0; sub < 2; ++sub) {
            const int key = kb + sub * 16 + ln;
#pragma unroll
            for (int r = 0; r < 8; ++r) {
              const int q = q0 + qs * 16 + r + (hi ? 8 : 0);
              if (key > q) s[qs][sub][r] = -__builtin_inff();
            }
          }
      }

      // ---- online softmax ----
#pragma unroll
      for (int qs = 0; qs < 2; ++qs) {
        float alpha[8], mnew[8];
#pragma unroll
        for (int r = 0; r < 8; ++r) {
          float t = fmaxf(s[qs][0][r], s[qs][1][r]);
          t = fmaxf(t, __shfl_xor(t, 8, 32));
          t = fmaxf(t, __shfl_xor(t, 4, 32));
          t = fmaxf(t, __shfl_xor(t, 2, 32));
          t = fmaxf(t, __shfl_xor(t, 1, 32));
          mnew[r]  = fmaxf(m[qs][r], t);
          alpha[r] = __expf(m[qs][r] - mnew[r]);
          m[qs][r] = mnew[r];
        }
#pragma unroll
        for (int r = 0; r < 8; ++r) {
          float p0 = __expf(s[qs][0][r] - mnew[r]);
          float p1 = __expf(s[qs][1][r] - mnew[r]);
          s[qs][0][r] = p0; s[qs][1][r] = p1;
          float t = p0 + p1;
          t += __shfl_xor(t, 8, 32);
          t += __shfl_xor(t, 4, 32);
          t += __shfl_xor(t, 2, 32);
          t += __shfl_xor(t, 1, 32);
          l[qs][r] = l[qs][r] * alpha[r] + t;
        }
#pragma unroll
        for (int t4 = 0; t4 < 4; ++t4)
#pragma unroll
          for (int r = 0; r < 8; ++r) acc[qs][t4][r] *= alpha[r];
      }

      // ---- P: C-layout -> LDS -> A-layout ----
      unsigned short* P = &sP[wave * QT2 * PSTR];
#pragma unroll
      for (int qs = 0; qs < 2; ++qs)
#pragma unroll
        for (int r = 0; r < 8; ++r) {
          const int q = qs * 16 + r + (hi ? 8 : 0);
          P[q * PSTR + ln]      = f2bf(s[qs][0][r]);
          P[q * PSTR + 16 + ln] = f2bf(s[qs][1][r]);
        }
      asm volatile("s_wait_dscnt 0" ::: "memory");
      v8u pa[2];
#pragma unroll
      for (int qs = 0; qs < 2; ++qs)
#pragma unroll
        for (int v = 0; v < 8; ++v) {
          const int k0 = 2 * v + (hi ? 16 : 0);
          pa[qs][v] = *(const unsigned int*)&P[(qs * 16 + ln) * PSTR + k0];
        }

      // ---- O += P V : V-B operands loaded once, reused for both q-subtiles ----
#pragma unroll
      for (int t4 = 0; t4 < 4; ++t4) {
        v8u vb;
#pragma unroll
        for (int v = 0; v < 8; ++v)
          vb[v] = *(const unsigned int*)&cV[(t4 * 16 + ln) * VSTR2 + 2 * v + (hi ? 16 : 0)];
        acc[0][t4] = WMMA_BF16(pa[0], vb, acc[0][t4]);
        acc[1][t4] = WMMA_BF16(pa[1], vb, acc[1][t4]);
      }
    }
  }

  // ---- epilogue ----
#pragma unroll
  for (int qs = 0; qs < 2; ++qs)
#pragma unroll
    for (int r = 0; r < 8; ++r) {
      const int q = q0 + qs * 16 + r + (hi ? 8 : 0);
      const float inv = 1.0f / l[qs][r];
      Op[(size_t)q * kD + 0  + ln] = acc[qs][0][r] * inv;
      Op[(size_t)q * kD + 16 + ln] = acc[qs][1][r] * inv;
      Op[(size_t)q * kD + 32 + ln] = acc[qs][2][r] * inv;
      Op[(size_t)q * kD + 48 + ln] = acc[qs][3][r] * inv;
    }
}

// ---------------- prepass: K f32 -> bf16 row-major ----------------
__global__ __launch_bounds__(256) void convert_k_kernel(
    const float* __restrict__ src, unsigned short* __restrict__ dst, int npairs) {
  int i = blockIdx.x * blockDim.x + threadIdx.x;
  if (i < npairs) {
    float2 f = ((const float2*)src)[i];
    ((unsigned int*)dst)[i] = pack2bf(f.x, f.y);
  }
}

// ---------------- prepass: V f32 [s][d] -> bf16 transposed [d][s] ----------------
__global__ __launch_bounds__(256) void transpose_v_kernel(
    const float* __restrict__ V, unsigned short* __restrict__ VT) {
  __shared__ unsigned short tile[64 * KSTR2];  // [s][d], padded
  const int stiles = kS / 64;
  const int bh = blockIdx.x / stiles;
  const int s0 = (blockIdx.x % stiles) * 64;
  const int t = threadIdx.x;
  {
    const int row = t >> 2, cq = (t & 3) * 16;
    const float* src = V + (size_t)bh * kS * kD + (size_t)(s0 + row) * kD + cq;
#pragma unroll
    for (int i = 0; i < 16; i += 2) {
      float2 f = *(const float2*)(src + i);
      *(unsigned int*)&tile[row * KSTR2 + cq + i] = pack2bf(f.x, f.y);
    }
  }
  __syncthreads();
  {
    const int d = t >> 2, sq = (t & 3) * 16;
    unsigned short* dst = VT + (size_t)bh * kD * kS + (size_t)d * kS + s0 + sq;
#pragma unroll
    for (int i = 0; i < 16; i += 2) {
      unsigned int u = (unsigned int)tile[(sq + i) * KSTR2 + d] |
                       ((unsigned int)tile[(sq + i + 1) * KSTR2 + d] << 16);
      *(unsigned int*)&dst[i] = u;
    }
  }
}

// ---------------- fallback (self-contained, QT=16) ----------------
constexpr int QT = 16;
constexpr int KSTR = kD + 2;
constexpr int VSTR = KC + 2;

__global__ __launch_bounds__(NW * 32) void fa2_wmma_kernel(
    const float* __restrict__ Qg, const float* __restrict__ Kg,
    const float* __restrict__ Vg, float* __restrict__ Og) {
  __shared__ unsigned short sK[KC * KSTR];
  __shared__ unsigned short sV[kD * VSTR];
  __shared__ unsigned short sP[NW * QT * PSTR];

  const int tid = threadIdx.x;
  const int wave = tid >> 5;
  const int lane = tid & 31;
  const int ln = lane & 15;
  const int hi = lane >> 4;

  const int qblocks = kS / (QT * NW);
  const int bh = blockIdx.x / qblocks;
  const int qb = (blockIdx.x % qblocks) * (QT * NW);
  const int q0 = qb + wave * QT;

  const float* Qp = Qg + (size_t)bh * kS * kD;
  const float* Kp = Kg + (size_t)bh * kS * kD;
  const float* Vp = Vg + (size_t)bh * kS * kD;
  float*       Op = Og + (size_t)bh * kS * kD;

  v8u qa0, qa1;
  {
    const float* qrow = Qp + (size_t)(q0 + ln) * kD;
#pragma unroll
    for (int v = 0; v < 8; ++v) {
      const int k0 = 2 * v + (hi ? 16 : 0);
      float2 f0 = *(const float2*)(qrow + k0);
      float2 f1 = *(const float2*)(qrow + k0 + 32);
      qa0[v] = pack2bf(f0.x * SCALE, f0.y * SCALE);
      qa1[v] = pack2bf(f1.x * SCALE, f1.y * SCALE);
    }
  }

  v8f acc0 = {}, acc1 = {}, acc2 = {}, acc3 = {};
  float m[8], l[8];
#pragma unroll
  for (int r = 0; r < 8; ++r) { m[r] = -__builtin_inff(); l[r] = 0.0f; }

  const int cmax = (qb + QT * NW - 1) / KC;
  for (int c = 0; c <= cmax; ++c) {
    const int kb = c * KC;
    __syncthreads();
    {
      const int row = tid >> 2;
      const int dq  = (tid & 3) * 16;
      const float* ks = Kp + (size_t)(kb + row) * kD + dq;
      const float* vs = Vp + (size_t)(kb + row) * kD + dq;
#pragma unroll
      for (int i = 0; i < 16; i += 4) {
        float4 f = *(const float4*)(ks + i);
        unsigned int* d32 = (unsigned int*)&sK[row * KSTR + dq + i];
        d32[0] = pack2bf(f.x, f.y);
        d32[1] = pack2bf(f.z, f.w);
        float4 g = *(const float4*)(vs + i);
        sV[(dq + i + 0) * VSTR + row] = f2bf(g.x);
        sV[(dq + i + 1) * VSTR + row] = f2bf(g.y);
        sV[(dq + i + 2) * VSTR + row] = f2bf(g.z);
        sV[(dq + i + 3) * VSTR + row] = f2bf(g.w);
      }
    }
    __syncthreads();
    if (kb > q0 + QT - 1) continue;

    v8f s[2];
#pragma unroll
    for (int sub = 0; sub < 2; ++sub) {
      v8u b0, b1;
      const int krow = sub * 16 + ln;
#pragma unroll
      for (int v = 0; v < 8; ++v) {
        const int d0 = 2 * v + (hi ? 16 : 0);
        b0[v] = *(const unsigned int*)&sK[krow * KSTR + d0];
        b1[v] = *(const unsigned int*)&sK[krow * KSTR + d0 + 32];
      }
      v8f z = {};
      z = WMMA_BF16(qa0, b0, z);
      z = WMMA_BF16(qa1, b1, z);
      s[sub] = z;
    }

    if (kb + KC > q0) {
#pragma unroll
      for (int sub = 0; sub < 2; ++sub) {
        const int key = kb + sub * 16 + ln;
#pragma unroll
        for (int r = 0; r < 8; ++r) {
          const int q = q0 + r + (hi ? 8 : 0);
          if (key > q) s[sub][r] = -__builtin_inff();
        }
      }
    }

    float alpha[8], mnew[8];
#pragma unroll
    for (int r = 0; r < 8; ++r) {
      float t = fmaxf(s[0][r], s[1][r]);
      t = fmaxf(t, __shfl_xor(t, 8, 32));
      t = fmaxf(t, __shfl_xor(t, 4, 32));
      t = fmaxf(t, __shfl_xor(t, 2, 32));
      t = fmaxf(t, __shfl_xor(t, 1, 32));
      mnew[r]  = fmaxf(m[r], t);
      alpha[r] = __expf(m[r] - mnew[r]);
      m[r] = mnew[r];
    }
#pragma unroll
    for (int r = 0; r < 8; ++r) {
      float p0 = __expf(s[0][r] - mnew[r]);
      float p1 = __expf(s[1][r] - mnew[r]);
      s[0][r] = p0; s[1][r] = p1;
      float t = p0 + p1;
      t += __shfl_xor(t, 8, 32);
      t += __shfl_xor(t, 4, 32);
      t += __shfl_xor(t, 2, 32);
      t += __shfl_xor(t, 1, 32);
      l[r] = l[r] * alpha[r] + t;
    }
#pragma unroll
    for (int r = 0; r < 8; ++r) {
      acc0[r] *= alpha[r]; acc1[r] *= alpha[r];
      acc2[r] *= alpha[r]; acc3[r] *= alpha[r];
    }

    unsigned short* P = &sP[wave * QT * PSTR];
#pragma unroll
    for (int r = 0; r < 8; ++r) {
      const int q = r + (hi ? 8 : 0);
      P[q * PSTR + ln]      = f2bf(s[0][r]);
      P[q * PSTR + 16 + ln] = f2bf(s[1][r]);
    }
    asm volatile("s_wait_dscnt 0" ::: "memory");
    v8u pa;
#pragma unroll
    for (int v = 0; v < 8; ++v) {
      const int k0 = 2 * v + (hi ? 16 : 0);
      pa[v] = *(const unsigned int*)&P[ln * PSTR + k0];
    }

    v8u vb;
#pragma unroll
    for (int v = 0; v < 8; ++v)
      vb[v] = *(const unsigned int*)&sV[(0 + ln) * VSTR + 2 * v + (hi ? 16 : 0)];
    acc0 = WMMA_BF16(pa, vb, acc0);
#pragma unroll
    for (int v = 0; v < 8; ++v)
      vb[v] = *(const unsigned int*)&sV[(16 + ln) * VSTR + 2 * v + (hi ? 16 : 0)];
    acc1 = WMMA_BF16(pa, vb, acc1);
#pragma unroll
    for (int v = 0; v < 8; ++v)
      vb[v] = *(const unsigned int*)&sV[(32 + ln) * VSTR + 2 * v + (hi ? 16 : 0)];
    acc2 = WMMA_BF16(pa, vb, acc2);
#pragma unroll
    for (int v = 0; v < 8; ++v)
      vb[v] = *(const unsigned int*)&sV[(48 + ln) * VSTR + 2 * v + (hi ? 16 : 0)];
    acc3 = WMMA_BF16(pa, vb, acc3);
  }

#pragma unroll
  for (int r = 0; r < 8; ++r) {
    const int q = q0 + r + (hi ? 8 : 0);
    const float inv = 1.0f / l[r];
    Op[(size_t)q * kD + 0  + ln] = acc0[r] * inv;
    Op[(size_t)q * kD + 16 + ln] = acc1[r] * inv;
    Op[(size_t)q * kD + 32 + ln] = acc2[r] * inv;
    Op[(size_t)q * kD + 48 + ln] = acc3[r] * inv;
  }
}

extern "C" void kernel_launch(void* const* d_in, const int* in_sizes, int n_in,
                              void* d_out, int out_size, void* d_ws, size_t ws_size,
                              hipStream_t stream) {
  (void)in_sizes; (void)n_in; (void)out_size;
  const float* Q = (const float*)d_in[0];
  const float* K = (const float*)d_in[1];
  const float* V = (const float*)d_in[2];
  float* O = (float*)d_out;

  const size_t elems  = (size_t)kB * kH * kS * kD;            // 4,194,304
  const size_t needed = 2 * elems * sizeof(unsigned short);   // ~16.8 MB

  if (ws_size >= needed) {
    unsigned short* wsK  = (unsigned short*)d_ws;
    unsigned short* wsVT = wsK + elems;
    const int npairs = (int)(elems / 2);
    convert_k_kernel<<<(npairs + 255) / 256, 256, 0, stream>>>(K, wsK, npairs);
    transpose_v_kernel<<<kB * kH * (kS / 64), 256, 0, stream>>>(V, wsVT);
    fa2_wmma_bf16_kernel<<<kB * kH * (kS / (QT2 * NW)), NW * 32, 0, stream>>>(Q, wsK, wsVT, O);
  } else {
    fa2_wmma_kernel<<<kB * kH * (kS / (QT * NW)), NW * 32, 0, stream>>>(Q, K, V, O);
  }
}
